// MultiHeadAttention_6983616823370
// MI455X (gfx1250) — matrix operand order repaired and checked by reading, not compiled, not run on hardware
//
#include <hip/hip_runtime.h>
#include <math.h>

// ---------------------------------------------------------------------------
// MI455X / gfx1250 multi-head attention forward, bf16 WMMA everywhere.
// D=768, H=12, Dh=64, B=2, S=2048.  All matmuls on v_wmma_f32_16x16x32_bf16.
// Q tile staged through the Tensor Data Mover (tensor_load_to_lds +
// s_wait_tensorcnt); P tile re-laid-out D->A through LDS (ds ops +
// s_wait_dscnt).
// ---------------------------------------------------------------------------

typedef __bf16 bf16;
typedef __attribute__((ext_vector_type(16))) __bf16 v16bf;
typedef __attribute__((ext_vector_type(8)))  __bf16 v8bf;
typedef __attribute__((ext_vector_type(8)))  float  v8f;
typedef __attribute__((ext_vector_type(4)))  unsigned int u32x4;
typedef __attribute__((ext_vector_type(8)))  int i32x8;
typedef __attribute__((ext_vector_type(4)))  int i32x4;

#define S_LEN   2048
#define D_MODEL 768
#define N_QKV   2304
#define NHEAD   12
#define DHEAD   64
#define M_TOT   4096   // B * S

#if defined(__has_builtin)
#if __has_builtin(__builtin_amdgcn_tensor_load_to_lds) && \
    __has_builtin(__builtin_amdgcn_s_wait_tensorcnt)
#define HAVE_TDM 1
#endif
#endif

// ---- WMMA fragment loaders (layouts per CDNA5 ISA 7.12.2) ------------------

// A-matrix 16x32 bf16, row-major source [.., K], ld in elements.
// lane L holds row M = L%16.  lanes 0-15: elems 0-7 -> K 0..7, elems 8-15 -> K 16..23
//                             lanes 16-31: +8 on both K ranges.
static __device__ __forceinline__ v16bf wmma_ld_a(const bf16* base, int ld, int row0, int k0) {
  const int lane = threadIdx.x & 31;
  const bf16* p = base + (size_t)(row0 + (lane & 15)) * ld + k0 + ((lane >> 4) << 3);
  v8bf lo = *(const v8bf*)(p);        // K = off .. off+7
  v8bf hi = *(const v8bf*)(p + 16);   // K = off+16 .. off+23
  v16bf r;
#pragma unroll
  for (int i = 0; i < 8; ++i) { r[i] = lo[i]; r[i + 8] = hi[i]; }
  return r;
}

// B-matrix 32x16 bf16 where the source is stored n-major [N, K] (ld = K stride),
// i.e. B[k][n] = src[n*ld + k].  lane L holds column N = L%16; lanes 0-15: elems
// 0-7 -> K 0..7, elems 8-15 -> K 16..23; lanes 16-31: +8 on both K ranges.
static __device__ __forceinline__ v16bf wmma_ld_b(const bf16* base, int ld, int n0, int k0) {
  const int lane = threadIdx.x & 31;
  const bf16* p = base + (size_t)(n0 + (lane & 15)) * ld + k0 + ((lane >> 4) << 3);
  v8bf lo = *(const v8bf*)(p);
  v8bf hi = *(const v8bf*)(p + 16);
  v16bf r;
#pragma unroll
  for (int i = 0; i < 8; ++i) { r[i] = lo[i]; r[i + 8] = hi[i]; }
  return r;
}

static __device__ __forceinline__ v8f wmma_bf16(v16bf a, v16bf b, v8f c) {
  // D = A x B + C, f32 accum, 16x16x32
  return __builtin_amdgcn_wmma_f32_16x16x32_bf16(false, a, false, b, (short)0, c,
                                                 false, false);
}

// Row-wise reductions over the 16 lanes holding one C/D row (xor masks < 16
// stay inside each wave32 half, which is exactly one M-group).
static __device__ __forceinline__ float rowmax16(float v) {
#pragma unroll
  for (int o = 8; o >= 1; o >>= 1) v = fmaxf(v, __shfl_xor(v, o, 32));
  return v;
}
static __device__ __forceinline__ float rowsum16(float v) {
#pragma unroll
  for (int o = 8; o >= 1; o >>= 1) v += __shfl_xor(v, o, 32);
  return v;
}

// ---- Tensor Data Mover: 2D bf16 tile load, global -> LDS -------------------
// Builds a D# descriptor (groups 0/1; groups 2/3 zero => <=2D tensor) and
// issues TENSOR_LOAD_TO_LDS.  All inputs must be wave-uniform.
static __device__ __forceinline__ void tdm_load_tile_bf16(
    unsigned lds_byte_off, const void* gptr,
    unsigned tile_rows, unsigned tile_cols, unsigned row_stride_elems) {
#if defined(HAVE_TDM)
  const unsigned long long ga = (unsigned long long)(uintptr_t)gptr;
  u32x4 g0;
  g0[0] = 1u;                                   // count=1, user descriptor
  g0[1] = lds_byte_off;                         // lds_addr
  g0[2] = (unsigned)ga;                         // global_addr[31:0]
  g0[3] = (unsigned)((ga >> 32) & 0x01FFFFFFu)  // global_addr[56:32]
          | 0x80000000u;                        // type = 2 ("image")
  i32x8 g1;
  g1[0] = 0x00010000;                           // data_size = 1 (2 bytes)
  g1[1] = (int)(tile_cols << 16);               // tensor_dim0[15:0]
  g1[2] = (int)(tile_rows << 16);               // tensor_dim1[15:0]
  g1[3] = (int)(tile_cols << 16);               // tile_dim0
  g1[4] = (int)tile_rows;                       // tile_dim1 (tile_dim2 = 0)
  g1[5] = (int)row_stride_elems;                // tensor_dim0_stride[31:0]
  g1[6] = 0;
  g1[7] = 0;
  i32x4 g2 = {0, 0, 0, 0};
  i32x4 g3 = {0, 0, 0, 0};
#if __has_include(<hip/amd_detail/amd_gfx1250_TDM.h>)
  i32x8 g4 = {0, 0, 0, 0, 0, 0, 0, 0};
  __builtin_amdgcn_tensor_load_to_lds(g0, g1, g2, g3, g4, 0);   // clang-23 form
#else
  __builtin_amdgcn_tensor_load_to_lds(g0, g1, g2, g3, 0);       // ROCm 7.2 form
#endif
#else
  (void)lds_byte_off; (void)gptr; (void)tile_rows; (void)tile_cols;
  (void)row_stride_elems;
#endif
}

// ---------------------------------------------------------------------------
// Kernel 0: fp32 -> bf16 conversion
// ---------------------------------------------------------------------------
__global__ void f32_to_bf16_kernel(const float* __restrict__ src,
                                   bf16* __restrict__ dst, int n) {
  int i = blockIdx.x * blockDim.x + threadIdx.x;
  if (i < n) dst[i] = (bf16)src[i];
}

// ---------------------------------------------------------------------------
// Kernel 1: QKV projection GEMM.  C[4096,2304] = q_bf @ W_qkv^T + b_qkv,
// scattered into Q [B,H,S,Dh] (pre-scaled by 1/8), K [B,H,S,Dh], V^T [B,H,Dh,S].
// One wave = 16x64 C tile.
// ---------------------------------------------------------------------------
__global__ __launch_bounds__(128) void qkv_gemm_kernel(
    const bf16* __restrict__ A,     // [4096,768]
    const bf16* __restrict__ W,     // [2304,768] (n-major: perfect B layout)
    const float* __restrict__ bias, // [2304]
    bf16* __restrict__ Qb, bf16* __restrict__ Kb, bf16* __restrict__ Vt) {
  const int lane = threadIdx.x & 31;
  const int wave = threadIdx.x >> 5;
  const int wid  = blockIdx.x * 4 + wave;       // 9216 waves total
  const int mt   = wid / 36;                    // 256 row tiles
  const int nt   = wid % 36;                    // 36  col tiles of 64
  const int row0 = mt * 16;
  const int col0 = nt * 64;

  v8f acc[4];
#pragma unroll
  for (int t = 0; t < 4; ++t) {
    const float bv = bias[col0 + t * 16 + (lane & 15)];
#pragma unroll
    for (int r = 0; r < 8; ++r) acc[t][r] = bv;
  }

  for (int kk = 0; kk < D_MODEL; kk += 32) {
    v16bf a = wmma_ld_a(A, D_MODEL, row0, kk);
    __builtin_prefetch(A + (size_t)(row0 + (lane & 15)) * D_MODEL + kk + 64, 0, 3);
#pragma unroll
    for (int t = 0; t < 4; ++t) {
      v16bf b = wmma_ld_b(W, D_MODEL, col0 + t * 16, kk);
      acc[t] = wmma_bf16(a, b, acc[t]);
    }
  }

  // Scatter. C/D layout: VGPR r, lane L -> row = r + 8*(L>>4), col = L&15.
#pragma unroll
  for (int t = 0; t < 4; ++t) {
    const int n     = col0 + t * 16 + (lane & 15);
    const int which = n / D_MODEL;   // 0=Q 1=K 2=V (uniform per 16-lane group)
    const int c     = n % D_MODEL;
    const int h     = c >> 6;
    const int dh    = c & 63;
#pragma unroll
    for (int r = 0; r < 8; ++r) {
      const int m  = row0 + r + ((lane >> 4) << 3);
      const int bi = m >> 11;        // / 2048
      const int s  = m & 2047;
      const int bh = bi * NHEAD + h;
      const float v = acc[t][r];
      if (which == 0)
        Qb[((size_t)(bh * S_LEN + s)) * DHEAD + dh] = (bf16)(v * 0.125f);
      else if (which == 1)
        Kb[((size_t)(bh * S_LEN + s)) * DHEAD + dh] = (bf16)v;
      else
        Vt[((size_t)(bh * DHEAD + dh)) * S_LEN + s] = (bf16)v;
    }
  }
}

// ---------------------------------------------------------------------------
// Kernel 2: flash attention.  One wave = 16 queries of one (b,h) against all
// 2048 keys, 32 keys per step.  Q tile staged via TDM, scores via 4 WMMAs,
// online softmax in f32 with cross-lane row reductions, P re-laid-out D->A
// via LDS, PV via 4 WMMAs against V^T (contiguous B-frags).
// ---------------------------------------------------------------------------
__global__ __launch_bounds__(128) void attn_kernel(
    const bf16* __restrict__ Qb,  // [BH, S, 64], pre-scaled by 1/sqrt(Dh)
    const bf16* __restrict__ Kb,  // [BH, S, 64]
    const bf16* __restrict__ Vt,  // [BH, 64, S]
    bf16* __restrict__ O) {       // [B, S, 768] bf16
  __shared__ __align__(16) bf16 ldsQ[4][16 * 64];
  __shared__ __align__(16) bf16 ldsP[4][16 * 32];

  const int lane = threadIdx.x & 31;
  const int wave = threadIdx.x >> 5;
  const int wid  = blockIdx.x * 4 + wave;       // 3072 waves
  const int bh   = wid >> 7;                    // / 128
  const int qt   = wid & 127;
  const int q0   = qt * 16;

  const bf16* Qh = Qb + (size_t)bh * S_LEN * DHEAD;
  const bf16* Kh = Kb + (size_t)bh * S_LEN * DHEAD;
  const bf16* Vh = Vt + (size_t)bh * DHEAD * S_LEN;

  // --- Stage the 16x64 Q tile into LDS with the Tensor Data Mover ---------
  bf16* Qs = &ldsQ[wave][0];
#if defined(HAVE_TDM)
  tdm_load_tile_bf16((unsigned)(uintptr_t)Qs, Qh + (size_t)q0 * DHEAD,
                     16u, 64u, 64u);
  __builtin_amdgcn_s_wait_tensorcnt(0);
#else
  // fallback: cooperative copy (each lane: one row-half of 32 bf16)
  {
    const int rr = lane & 15, hh = lane >> 4;
    const v8bf* src = (const v8bf*)(Qh + (size_t)(q0 + rr) * DHEAD + hh * 32);
    v8bf* dst = (v8bf*)(Qs + rr * DHEAD + hh * 32);
    dst[0] = src[0];
    dst[1] = src[1];
    dst[2] = src[2];
    dst[3] = src[3];
  }
  asm volatile("s_wait_dscnt 0x0" ::: "memory");
#endif
  const v16bf aq0 = wmma_ld_a(Qs, DHEAD, 0, 0);
  const v16bf aq1 = wmma_ld_a(Qs, DHEAD, 0, 32);

  float m_r[8], l_r[8];
  v8f acc[4];
#pragma unroll
  for (int r = 0; r < 8; ++r) { m_r[r] = -INFINITY; l_r[r] = 0.f; }
#pragma unroll
  for (int t = 0; t < 4; ++t)
#pragma unroll
    for (int r = 0; r < 8; ++r) acc[t][r] = 0.f;

  bf16* P = &ldsP[wave][0];

  for (int j = 0; j < S_LEN; j += 32) {
    // scores tile 16 x 32 (two 16x16 D frags), K-dim = Dh = 64 = 2 WMMA steps
    v8f s0 = {}, s1 = {};
    s0 = wmma_bf16(aq0, wmma_ld_b(Kh, DHEAD, j, 0),       s0);
    s0 = wmma_bf16(aq1, wmma_ld_b(Kh, DHEAD, j, 32),      s0);
    s1 = wmma_bf16(aq0, wmma_ld_b(Kh, DHEAD, j + 16, 0),  s1);
    s1 = wmma_bf16(aq1, wmma_ld_b(Kh, DHEAD, j + 16, 32), s1);
    __builtin_prefetch(Kh + (size_t)(j + 32 + (lane & 15)) * DHEAD, 0, 3);

    // online softmax, per accumulator row r (row = r + 8*(lane>>4))
#pragma unroll
    for (int r = 0; r < 8; ++r) {
      float mx = rowmax16(fmaxf(s0[r], s1[r]));
      const float mnew  = fmaxf(m_r[r], mx);
      const float alpha = __expf(m_r[r] - mnew);
      const float p0 = __expf(s0[r] - mnew);
      const float p1 = __expf(s1[r] - mnew);
      l_r[r] = l_r[r] * alpha + rowsum16(p0 + p1);
      m_r[r] = mnew;
#pragma unroll
      for (int t = 0; t < 4; ++t) acc[t][r] *= alpha;
      const int row = r + ((lane >> 4) << 3);
      P[row * 32 + (lane & 15)]      = (bf16)p0;  // D-layout -> row-major LDS
      P[row * 32 + 16 + (lane & 15)] = (bf16)p1;
    }
    // wave-internal LDS RAW: split-counter wait (CDNA5), then A-layout reload
    asm volatile("s_wait_dscnt 0x0" ::: "memory");
    const v16bf pf = wmma_ld_a(P, 32, 0, 0);

    // PV: acc[16 x 64] += P[16 x 32] @ V[32 x 64]  (V^T gives contiguous B)
#pragma unroll
    for (int t = 0; t < 4; ++t) {
      v16bf bv = wmma_ld_b(Vh, S_LEN, t * 16, j);
      acc[t] = wmma_bf16(pf, bv, acc[t]);
    }
  }

  // epilogue: normalize and write [B,S,768] bf16
  const int bi = bh / NHEAD;
  const int h  = bh % NHEAD;
#pragma unroll
  for (int t = 0; t < 4; ++t) {
#pragma unroll
    for (int r = 0; r < 8; ++r) {
      const int s = q0 + r + ((lane >> 4) << 3);
      const float ov = acc[t][r] / l_r[r];
      const size_t g =
          (size_t)(bi * S_LEN + s) * D_MODEL + h * DHEAD + t * 16 + (lane & 15);
      O[g] = (bf16)ov;
    }
  }
}

// ---------------------------------------------------------------------------
// Kernel 3: output projection.  out[4096,768] = attn_bf @ W_out^T + b_out (f32)
// ---------------------------------------------------------------------------
__global__ __launch_bounds__(128) void out_gemm_kernel(
    const bf16* __restrict__ A,     // [4096,768]
    const bf16* __restrict__ W,     // [768,768]
    const float* __restrict__ bias, // [768]
    float* __restrict__ out) {      // [4096,768] f32
  const int lane = threadIdx.x & 31;
  const int wave = threadIdx.x >> 5;
  const int wid  = blockIdx.x * 4 + wave;   // 3072 waves
  const int mt   = wid / 12;
  const int nt   = wid % 12;
  const int row0 = mt * 16;
  const int col0 = nt * 64;

  v8f acc[4];
#pragma unroll
  for (int t = 0; t < 4; ++t) {
    const float bv = bias[col0 + t * 16 + (lane & 15)];
#pragma unroll
    for (int r = 0; r < 8; ++r) acc[t][r] = bv;
  }

  for (int kk = 0; kk < D_MODEL; kk += 32) {
    v16bf a = wmma_ld_a(A, D_MODEL, row0, kk);
    __builtin_prefetch(A + (size_t)(row0 + (lane & 15)) * D_MODEL + kk + 64, 0, 3);
#pragma unroll
    for (int t = 0; t < 4; ++t) {
      v16bf b = wmma_ld_b(W, D_MODEL, col0 + t * 16, kk);
      acc[t] = wmma_bf16(a, b, acc[t]);
    }
  }

#pragma unroll
  for (int t = 0; t < 4; ++t) {
    const int n = col0 + t * 16 + (lane & 15);
#pragma unroll
    for (int r = 0; r < 8; ++r) {
      const int m = row0 + r + ((lane >> 4) << 3);
      out[(size_t)m * D_MODEL + n] = acc[t][r];
    }
  }
}

// ---------------------------------------------------------------------------
extern "C" void kernel_launch(void* const* d_in, const int* in_sizes, int n_in,
                              void* d_out, int out_size, void* d_ws, size_t ws_size,
                              hipStream_t stream) {
  (void)in_sizes; (void)n_in; (void)out_size; (void)ws_size;
  const float* q     = (const float*)d_in[0];
  // d_in[1], d_in[2] (k, v) are ignored by the reference module.
  const float* W_qkv = (const float*)d_in[3];
  const float* b_qkv = (const float*)d_in[4];
  const float* W_out = (const float*)d_in[5];
  const float* b_out = (const float*)d_in[6];
  float* out = (float*)d_out;

  // workspace carve-up (bf16), ~30 MB total, every block 16B-aligned
  bf16* qbf  = (bf16*)d_ws;                 // 4096*768
  bf16* wqkv = qbf  + M_TOT * D_MODEL;      // 2304*768
  bf16* wout = wqkv + N_QKV * D_MODEL;      // 768*768
  bf16* Qb   = wout + D_MODEL * D_MODEL;    // 24*2048*64 (scaled by 1/8)
  bf16* Kb   = Qb   + 24 * S_LEN * DHEAD;
  bf16* Vt   = Kb   + 24 * S_LEN * DHEAD;   // transposed [BH, Dh, S]
  bf16* aob  = Vt   + 24 * S_LEN * DHEAD;   // attention output, bf16 [4096,768]

  const int n1 = M_TOT * D_MODEL;
  const int n2 = N_QKV * D_MODEL;
  const int n3 = D_MODEL * D_MODEL;
  f32_to_bf16_kernel<<<(n1 + 255) / 256, 256, 0, stream>>>(q, qbf, n1);
  f32_to_bf16_kernel<<<(n2 + 255) / 256, 256, 0, stream>>>(W_qkv, wqkv, n2);
  f32_to_bf16_kernel<<<(n3 + 255) / 256, 256, 0, stream>>>(W_out, wout, n3);

  // 9216 waves / 4 per block
  qkv_gemm_kernel<<<2304, 128, 0, stream>>>(qbf, wqkv, b_qkv, Qb, Kb, Vt);
  // 3072 waves / 4 per block
  attn_kernel<<<768, 128, 0, stream>>>(Qb, Kb, Vt, aob);
  // 3072 waves / 4 per block
  out_gemm_kernel<<<768, 128, 0, stream>>>(aob, wout, b_out, out);
}
